// ConventionalAttention_72902774882840
// MI455X (gfx1250) — matrix operand order repaired and checked by reading, not compiled, not run on hardware
//
#include <hip/hip_runtime.h>
#include <hip/hip_bf16.h>
#include <math.h>

typedef __attribute__((ext_vector_type(16))) __bf16 bf16x16;
typedef __attribute__((ext_vector_type(8)))  __bf16 bf16x8;
typedef __attribute__((ext_vector_type(4)))  __bf16 bf16x4;
typedef __attribute__((ext_vector_type(8)))  float  f32x8;
typedef __attribute__((ext_vector_type(4)))  float  f32x4;

#define BATCH 2
#define HEADS 16
#define BH    (BATCH * HEADS)
#define SEQ   2048
#define DIM   128
#define QB    64           // queries per block (4 waves x 16)
#define NTHREADS 128
#define NELEM ((size_t)BH * SEQ * DIM)

// 1/sqrt(128) * log2(e): softmax in exp2 domain, scale folded into Q
#define QK_SCALE_LOG2 (0.08838834764831845f * 1.44269504088896340736f)

// ---------- wave32 cross-lane helpers ----------
static __device__ __forceinline__ float xor16(float x) {          // swap with lane^16
  union { float f; int i; } v; v.f = x;
  v.i = __builtin_amdgcn_ds_swizzle(v.i, 0x401F);                 // xor 0x10, and 0x1f
  return v.f;
}
static __device__ __forceinline__ float bcast(float x, int srcLane) {
  union { float f; int i; } v; v.f = x;
  v.i = __builtin_amdgcn_ds_bpermute(srcLane << 2, v.i);
  return v.f;
}
static __device__ __forceinline__ float fast_exp2(float x) {
#if __has_builtin(__builtin_amdgcn_exp2f)
  return __builtin_amdgcn_exp2f(x);
#else
  return exp2f(x);
#endif
}

// ---------- CDNA5 async-to-LDS (ASYNCcnt) ----------
static __device__ __forceinline__ void async_b128(unsigned lds_off, const void* gsrc) {
  unsigned long long ga = (unsigned long long)gsrc;
  asm volatile("global_load_async_to_lds_b128 %0, %1, off"
               :: "v"(lds_off), "v"(ga) : "memory");
}
static __device__ __forceinline__ void wait_async0() {
  asm volatile("s_wait_asynccnt 0x0" ::: "memory");
}
static __device__ __forceinline__ unsigned lds_off(const void* p) {
  return (unsigned)(unsigned long long)p;   // flat LDS addr: low 32 bits = LDS offset
}

// ---------- pre-pass 1: Q (pre-scaled) and K -> bf16, layout unchanged ----------
__global__ __launch_bounds__(256)
void cvt_qk_kernel(const float* __restrict__ Q, const float* __restrict__ K,
                   __bf16* __restrict__ qo, __bf16* __restrict__ ko) {
  size_t i = (size_t)blockIdx.x * blockDim.x + threadIdx.x;       // one float4 each
  f32x4 q = ((const f32x4*)Q)[i];
  f32x4 k = ((const f32x4*)K)[i];
  bf16x4 qv, kv;
#pragma unroll
  for (int e = 0; e < 4; ++e) {
    qv[e] = (__bf16)(q[e] * QK_SCALE_LOG2);
    kv[e] = (__bf16)k[e];
  }
  ((bf16x4*)qo)[i] = qv;
  ((bf16x4*)ko)[i] = kv;
}

// ---------- pre-pass 2: V -> bf16 transposed Vt[bh][d][s] (LDS-tiled) ----------
#define TPAD 40
__global__ __launch_bounds__(256)
void cvt_vt_kernel(const float* __restrict__ V, __bf16* __restrict__ vt) {
  __shared__ __bf16 tile[DIM][TPAD];
  const int t  = threadIdx.x;
  const int bh = blockIdx.x / (SEQ / 32);
  const int s0 = (blockIdx.x % (SEQ / 32)) * 32;
  const float* Vb  = V  + (size_t)bh * SEQ * DIM;
  __bf16*      vtb = vt + (size_t)bh * SEQ * DIM;

  const int sl = t >> 3;
  const int db = (t & 7) << 4;
  const float* p = Vb + (size_t)(s0 + sl) * DIM + db;
#pragma unroll
  for (int e = 0; e < 16; e += 4) {
    f32x4 v = *(const f32x4*)(p + e);
    tile[db + e + 0][sl] = (__bf16)v[0];
    tile[db + e + 1][sl] = (__bf16)v[1];
    tile[db + e + 2][sl] = (__bf16)v[2];
    tile[db + e + 3][sl] = (__bf16)v[3];
  }
  __syncthreads();
  const int d = t >> 1;
  const int h = (t & 1) << 4;
  bf16x8 a = *(const bf16x8*)&tile[d][h];
  bf16x8 b = *(const bf16x8*)&tile[d][h + 8];
  __bf16* op = vtb + (size_t)d * SEQ + s0 + h;
  *(bf16x8*)op       = a;
  *(bf16x8*)(op + 8) = b;
}

// ---------- shared softmax / online-update (returns P in A-operand layout) ----------
template <bool MASKED>
static __device__ __forceinline__ bf16x16 softmax_update(
    f32x8& s0, f32x8& s1, f32x8 (&acc)[8], float& m, float& l,
    int kb, int qrow, int hi8) {
  float mloc = -INFINITY;
#pragma unroll
  for (int j = 0; j < 8; ++j) {
    float v0 = s0[j], v1 = s1[j];
    if constexpr (MASKED) {
      int k0 = kb + j + hi8;
      v0 = (k0 <= qrow) ? v0 : -INFINITY;
      v1 = (k0 + 16 <= qrow) ? v1 : -INFINITY;
      s0[j] = v0; s1[j] = v1;
    }
    mloc = fmaxf(mloc, fmaxf(v0, v1));
  }
  mloc = fmaxf(mloc, xor16(mloc));
  float mnew  = fmaxf(m, mloc);
  float alpha = fast_exp2(m - mnew);

  float sum = 0.0f;
  bf16x16 pA;
#pragma unroll
  for (int j = 0; j < 8; ++j) {
    float p0 = fast_exp2(s0[j] - mnew);
    float p1 = fast_exp2(s1[j] - mnew);
    sum += p0 + p1;
    pA[j]     = (__bf16)p0;
    pA[8 + j] = (__bf16)p1;
  }
  sum += xor16(sum);
  l = alpha * l + sum;
  m = mnew;
#pragma unroll
  for (int j = 0; j < 8; ++j) {
    float aj = bcast(alpha, j + hi8);
#pragma unroll
    for (int t = 0; t < 8; ++t) acc[t][j] *= aj;
  }
  return pA;
}

// ---------- stage one 32-key K tile + Vt tile into LDS, fragment-major ----------
// K granule g = ((tau*4+c)*2+seg)*32 + lane ; V granule g = ((t*2)+seg)*32 + lane
static __device__ __forceinline__ void stage_tiles(
    const __bf16* __restrict__ Kb, const __bf16* __restrict__ VTb, int kb,
    unsigned kbase, unsigned vbase, int tid) {
#pragma unroll
  for (int j = 0; j < 4; ++j) {
    int g = tid * 4 + j;
    int ln = g & 31, rest = g >> 5;
    int seg = rest & 1, c = (rest >> 1) & 3, tau = rest >> 3;
    int row = (ln & 15) + (tau << 4);
    int col = c * 32 + ((ln >> 4) << 4) + (seg << 3);
    async_b128(kbase + (unsigned)g * 16, Kb + (size_t)(kb + row) * DIM + col);
  }
#pragma unroll
  for (int j = 0; j < 4; ++j) {
    int g = tid * 4 + j;
    int ln = g & 31, rest = g >> 5;
    int seg = rest & 1, tt = rest >> 1;
    int d = tt * 16 + (ln & 15);
    int ko = ((ln >> 4) << 3) + (seg << 4);
    async_b128(vbase + (unsigned)g * 16, VTb + (size_t)d * SEQ + kb + ko);
  }
}

// ---------- one 32-key step, operands from LDS fragments (conflict-free) ----------
template <bool MASKED>
static __device__ __forceinline__ void attn_step_lds(
    int kb, const __bf16* __restrict__ kls, const __bf16* __restrict__ vls,
    const bf16x16 (&qB)[4], f32x8 (&acc)[8], float& m, float& l,
    int lane, int hi8, int qrow) {
  f32x8 s0, s1;
#pragma unroll
  for (int j = 0; j < 8; ++j) { s0[j] = 0.0f; s1[j] = 0.0f; }
#pragma unroll
  for (int c = 0; c < 4; ++c) {
    bf16x8 l0 = *(const bf16x8*)(kls + (size_t)(((c)     * 2 + 0) * 32 + lane) * 8);
    bf16x8 h0 = *(const bf16x8*)(kls + (size_t)(((c)     * 2 + 1) * 32 + lane) * 8);
    bf16x8 l1 = *(const bf16x8*)(kls + (size_t)(((4 + c) * 2 + 0) * 32 + lane) * 8);
    bf16x8 h1 = *(const bf16x8*)(kls + (size_t)(((4 + c) * 2 + 1) * 32 + lane) * 8);
    bf16x16 a0, a1;
#pragma unroll
    for (int e = 0; e < 8; ++e) {
      a0[e] = l0[e]; a0[8 + e] = h0[e];
      a1[e] = l1[e]; a1[8 + e] = h1[e];
    }
    s0 = __builtin_amdgcn_wmma_f32_16x16x32_bf16(false, a0, false, qB[c],
                                                 (short)0, s0, false, false);
    s1 = __builtin_amdgcn_wmma_f32_16x16x32_bf16(false, a1, false, qB[c],
                                                 (short)0, s1, false, false);
  }
  bf16x16 pA = softmax_update<MASKED>(s0, s1, acc, m, l, kb, qrow, hi8);
#pragma unroll
  for (int t = 0; t < 8; ++t) {
    bf16x8 lo = *(const bf16x8*)(vls + (size_t)((t * 2 + 0) * 32 + lane) * 8);
    bf16x8 hi = *(const bf16x8*)(vls + (size_t)((t * 2 + 1) * 32 + lane) * 8);
    bf16x16 vB;
#pragma unroll
    for (int e = 0; e < 8; ++e) { vB[e] = lo[e]; vB[8 + e] = hi[e]; }
    acc[t] = __builtin_amdgcn_wmma_f32_16x16x32_bf16(false, pA, false, vB,
                                                     (short)0, acc[t], false, false);
  }
}

// ---------- main kernel: async double-buffered LDS staging ----------
__global__ __launch_bounds__(NTHREADS)
void fa_fwd_lds(const __bf16* __restrict__ Qbf, const __bf16* __restrict__ Kbf,
                const __bf16* __restrict__ VTbf, float* __restrict__ O) {
  __shared__ __align__(16) __bf16 kbuf[2][32 * DIM];   // 8KB each
  __shared__ __align__(16) __bf16 vbuf[2][32 * DIM];   // 8KB each

  const int tid   = threadIdx.x;
  const int lane  = tid & 31;
  const int qblks = SEQ / QB;
  const int bh    = blockIdx.x / qblks;
  const int qblk  = blockIdx.x % qblks;
  // wave-uniform scalars pinned to SGPRs -> scalar branches, EXEC untouched at WMMA
  const int qbase = __builtin_amdgcn_readfirstlane(qblk * QB + ((tid >> 5) << 4));

  const size_t bhoff = (size_t)bh * SEQ * DIM;
  const int hi8  = (lane >> 4) << 3;
  const int bseg = (lane >> 4) << 4;
  const int qrow = qbase + (lane & 15);

  const __bf16* Kb  = Kbf  + bhoff;
  const __bf16* VTb = VTbf + bhoff;

  // Q^T B-operand in registers (pre-scaled bf16)
  bf16x16 qB[4];
  const __bf16* qRow = Qbf + bhoff + (size_t)qrow * DIM;
#pragma unroll
  for (int c = 0; c < 4; ++c) qB[c] = *(const bf16x16*)(qRow + c * 32 + bseg);

  f32x8 acc[8];
#pragma unroll
  for (int t = 0; t < 8; ++t)
#pragma unroll
    for (int j = 0; j < 8; ++j) acc[t][j] = 0.0f;
  float m = -INFINITY, l = 0.0f;

  const unsigned kb0 = lds_off(&kbuf[0][0]), kb1 = lds_off(&kbuf[1][0]);
  const unsigned vb0 = lds_off(&vbuf[0][0]), vb1 = lds_off(&vbuf[1][0]);

  const int kend_blk  = qblk * QB + QB;     // all waves iterate same range
  const int kend_wave = qbase + 16;         // SGPR (from readfirstlane'd qbase)
  const int kfull     = qbase & ~31;        // SGPR

  stage_tiles(Kb, VTb, 0, kb0, vb0, tid);
  wait_async0();
  __syncthreads();

  for (int kb = 0; kb < kend_blk; kb += 32) {
    const int cur = (kb >> 5) & 1;
    if (kb + 32 < kend_blk)                 // prefetch next tile while computing
      stage_tiles(Kb, VTb, kb + 32, cur ? kb0 : kb1, cur ? vb0 : vb1, tid);

    if (kb < kfull)
      attn_step_lds<false>(kb, &kbuf[cur][0], &vbuf[cur][0], qB, acc, m, l,
                           lane, hi8, qrow);
    else if (kb < kend_wave)
      attn_step_lds<true>(kb, &kbuf[cur][0], &vbuf[cur][0], qB, acc, m, l,
                          lane, hi8, qrow);

    wait_async0();                          // own async writes to next buffer done
    __syncthreads();                        // everyone done writing next / reading cur
  }

#if __has_builtin(__builtin_amdgcn_rcpf)
  float linv = __builtin_amdgcn_rcpf(l);
#else
  float linv = 1.0f / l;
#endif
  float* Ob = O + bhoff;
#pragma unroll
  for (int j = 0; j < 8; ++j) {
    float lj = bcast(linv, j + hi8);
    int q = qbase + j + hi8;
    float* op = Ob + (size_t)q * DIM + (lane & 15);
#pragma unroll
    for (int t = 0; t < 8; ++t) op[t * 16] = acc[t][j] * lj;
  }
}

// ---------- fallback: direct f32 path (no workspace) ----------
static __device__ __forceinline__ bf16x16 load_atile_f32(const float* __restrict__ rowPtr,
                                                         int c0, int hi8) {
  const float* p0 = rowPtr + c0 + 2 * hi8;
  const float* p1 = rowPtr + c0 + 8 + 2 * hi8;
  f32x4 a0 = *(const f32x4*)(p0);
  f32x4 a1 = *(const f32x4*)(p0 + 4);
  f32x4 b0 = *(const f32x4*)(p1);
  f32x4 b1 = *(const f32x4*)(p1 + 4);
  bf16x16 t;
#pragma unroll
  for (int e = 0; e < 4; ++e) {
    t[e]      = (__bf16)a0[e];
    t[4 + e]  = (__bf16)a1[e];
    t[8 + e]  = (__bf16)b0[e];
    t[12 + e] = (__bf16)b1[e];
  }
  return t;
}

template <bool MASKED>
static __device__ __forceinline__ void attn_step_f32(
    int kb, const float* __restrict__ Kf, const float* __restrict__ Vf,
    const bf16x16 (&qB)[4], f32x8 (&acc)[8], float& m, float& l,
    int lane, int hi8, int bseg, int qrow) {
  f32x8 s0, s1;
#pragma unroll
  for (int j = 0; j < 8; ++j) { s0[j] = 0.0f; s1[j] = 0.0f; }
#pragma unroll
  for (int c = 0; c < 4; ++c) {
    const float* r0 = Kf + (size_t)(kb + (lane & 15)) * DIM;
    bf16x16 a0 = load_atile_f32(r0, c * 32, hi8);
    bf16x16 a1 = load_atile_f32(r0 + (size_t)16 * DIM, c * 32, hi8);
    s0 = __builtin_amdgcn_wmma_f32_16x16x32_bf16(false, a0, false, qB[c],
                                                 (short)0, s0, false, false);
    s1 = __builtin_amdgcn_wmma_f32_16x16x32_bf16(false, a1, false, qB[c],
                                                 (short)0, s1, false, false);
  }
  bf16x16 pA = softmax_update<MASKED>(s0, s1, acc, m, l, kb, qrow, hi8);
#pragma unroll
  for (int t = 0; t < 8; ++t) {
    const float* p = Vf + (size_t)(kb + hi8) * DIM + t * 16 + (lane & 15);
    bf16x16 vB;
#pragma unroll
    for (int e = 0; e < 16; ++e) vB[e] = (__bf16)p[(size_t)(e + (e >= 8 ? 8 : 0)) * DIM];
    acc[t] = __builtin_amdgcn_wmma_f32_16x16x32_bf16(false, pA, false, vB,
                                                     (short)0, acc[t], false, false);
  }
}

__global__ __launch_bounds__(NTHREADS)
void fa_fwd_f32(const float* __restrict__ Qf, const float* __restrict__ Kf,
                const float* __restrict__ Vf, float* __restrict__ O) {
  const int tid   = threadIdx.x;
  const int lane  = tid & 31;
  const int qblks = SEQ / QB;
  const int bh    = blockIdx.x / qblks;
  const int qblk  = blockIdx.x % qblks;
  const int qbase = __builtin_amdgcn_readfirstlane(qblk * QB + ((tid >> 5) << 4));

  const size_t bhoff = (size_t)bh * SEQ * DIM;
  const int hi8  = (lane >> 4) << 3;
  const int bseg = (lane >> 4) << 4;
  const int qrow = qbase + (lane & 15);

  bf16x16 qB[4];
  const float* qRow = Qf + bhoff + (size_t)qrow * DIM;
#pragma unroll
  for (int c = 0; c < 4; ++c) {
    const float* p = qRow + c * 32 + bseg;
    bf16x16 t;
#pragma unroll
    for (int e = 0; e < 16; e += 4) {
      f32x4 v = *(const f32x4*)(p + e);
      t[e]   = (__bf16)(v[0] * QK_SCALE_LOG2);
      t[e+1] = (__bf16)(v[1] * QK_SCALE_LOG2);
      t[e+2] = (__bf16)(v[2] * QK_SCALE_LOG2);
      t[e+3] = (__bf16)(v[3] * QK_SCALE_LOG2);
    }
    qB[c] = t;
  }

  f32x8 acc[8];
#pragma unroll
  for (int t = 0; t < 8; ++t)
#pragma unroll
    for (int j = 0; j < 8; ++j) acc[t][j] = 0.0f;
  float m = -INFINITY, l = 0.0f;

  const int kend  = qbase + 16;
  const int kfull = qbase & ~31;
  int kb = 0;
  for (; kb < kfull; kb += 32)
    attn_step_f32<false>(kb, Qf == nullptr ? nullptr : Kf + bhoff, Vf + bhoff,
                         qB, acc, m, l, lane, hi8, bseg, qrow);
  for (; kb < kend; kb += 32)
    attn_step_f32<true>(kb, Kf + bhoff, Vf + bhoff, qB, acc, m, l,
                        lane, hi8, bseg, qrow);

#if __has_builtin(__builtin_amdgcn_rcpf)
  float linv = __builtin_amdgcn_rcpf(l);
#else
  float linv = 1.0f / l;
#endif
  float* Ob = O + bhoff;
#pragma unroll
  for (int j = 0; j < 8; ++j) {
    float lj = bcast(linv, j + hi8);
    int q = qbase + j + hi8;
    float* op = Ob + (size_t)q * DIM + (lane & 15);
#pragma unroll
    for (int t = 0; t < 8; ++t) op[t * 16] = acc[t][j] * lj;
  }
}

extern "C" void kernel_launch(void* const* d_in, const int* in_sizes, int n_in,
                              void* d_out, int out_size, void* d_ws, size_t ws_size,
                              hipStream_t stream) {
  const float* Q = (const float*)d_in[0];
  const float* K = (const float*)d_in[1];
  const float* V = (const float*)d_in[2];
  float* O = (float*)d_out;
  (void)in_sizes; (void)n_in; (void)out_size;

  const size_t need = 3 * NELEM * sizeof(__bf16);   // 48 MB
  dim3 grid(BH * (SEQ / QB));
  dim3 block(NTHREADS);

  if (ws_size >= need) {
    __bf16* qbf = (__bf16*)d_ws;
    __bf16* kbf = qbf + NELEM;
    __bf16* vt  = kbf + NELEM;
    cvt_qk_kernel<<<dim3((unsigned)(NELEM / 4 / 256)), dim3(256), 0, stream>>>(Q, K, qbf, kbf);
    cvt_vt_kernel<<<dim3(BH * (SEQ / 32)), dim3(256), 0, stream>>>(V, vt);
    fa_fwd_lds<<<grid, block, 0, stream>>>(qbf, kbf, vt, O);
  } else {
    fa_fwd_f32<<<grid, block, 0, stream>>>(Q, K, V, O);
  }
}
